// CrossModalMDTA_54614804136421
// MI455X (gfx1250) — matrix operand order repaired and checked by reading, not compiled, not run on hardware
//
#include <hip/hip_runtime.h>

// ---------------------------------------------------------------------------
// CrossModal MDTA for MI455X (gfx1250, wave32).
// GEMM-shaped work (1x1 convs, channel-attention GEMMs) -> v_wmma_f32_16x16x32_bf16
// (bf16 in, f32 accumulate). X tiles staged with GLOBAL_LOAD_ASYNC_TO_LDS_B128
// (ASYNCcnt-tracked, double buffered). Memory-bound parts stay f32 VALU.
// ---------------------------------------------------------------------------

typedef __attribute__((ext_vector_type(16))) __bf16 v16bf;
typedef __attribute__((ext_vector_type(8)))  float  v8f;

union FragBF {
  v16bf v;
  __bf16 e[16];
  unsigned u[8];
};
union FragF {
  v8f v;
  float e[8];
};

#define NSP 16384   // H*W
#define CCH 192     // channels
#define NB  8       // batch

// CDNA5 async global->LDS copy (16B), tracked by ASYNCcnt.
__device__ __forceinline__ void async_cp_b128(const float* gsrc, const float* ldst) {
  unsigned loff = (unsigned)(uintptr_t)ldst;   // low 32 bits of flat addr = LDS offset
  asm volatile("global_load_async_to_lds_b128 %0, %1, off"
               :: "v"(loff), "v"(gsrc)
               : "memory");
}
template <int N>
__device__ __forceinline__ void wait_asynccnt() {
  asm volatile("s_wait_asynccnt %0" :: "i"(N) : "memory");
}

// ---------------------------------------------------------------------------
// Pointwise (1x1) conv as GEMM: Y[b, co, n] = sum_ci W[co,ci] * X[b,ci,n]
// grid: (N/128, Co/64, B), block 256 (8 waves). Each wave owns one 16-col
// block and 4 stacked 16-row M tiles (64x128 per workgroup).
// Weights in LDS as bf16 (loaded once); X tiles double-buffered in LDS as f32
// via async-to-LDS, converted to bf16 at fragment build.
// ---------------------------------------------------------------------------
__global__ __launch_bounds__(256)
void pw_conv_bf16(const float* __restrict__ X, const float* __restrict__ W,
                  float* __restrict__ Y, int Ci, int N) {
  const int ntile = blockIdx.x;
  const int mtile = blockIdx.y;
  const int Co    = gridDim.y * 64;
  const int b     = blockIdx.z;
  const int tid   = threadIdx.x;
  const int lane  = tid & 31;
  const int wave  = tid >> 5;          // n-block 0..7
  const int lm    = lane & 15;
  const int hf    = lane >> 4;

  __shared__ __bf16 Wl[64 * 196];      // [64 rows][Ci<=192], padded stride
  __shared__ float  Xf[2 * 32 * 132];  // double-buffered tile [k][n], stride 132

  const float* Xb = X + (size_t)b * Ci * N;
  float*       Yb = Y + (size_t)b * Co * N;

  // stage weight block [64 x Ci] once (bf16)
  for (int i = tid; i < 64 * Ci; i += 256) {
    int r = i / Ci, c = i - r * Ci;
    Wl[r * 196 + c] = (__bf16)W[(size_t)(mtile * 64 + r) * Ci + c];
  }

  // async fill of one 32x128 K-tile: 1024 16B chunks, 4 per thread
  auto fill = [&](int buf, int kt) {
    const float* gsrc = Xb + (size_t)(kt * 32) * N + ntile * 128;
    const float* ldst = Xf + buf * (32 * 132);
#pragma unroll
    for (int c = 0; c < 4; ++c) {
      int idx = tid + c * 256;          // 0..1023
      int row = idx >> 5;               // 0..31
      int col = (idx & 31) * 4;         // 0..124
      async_cp_b128(gsrc + (size_t)row * N + col, ldst + row * 132 + col);
    }
  };

  FragF acc[4];
#pragma unroll
  for (int m = 0; m < 4; ++m)
#pragma unroll
    for (int r = 0; r < 8; ++r) acc[m].e[r] = 0.f;

  const int nk = Ci / 32;               // 6
  fill(0, 0);                           // prologue: ASYNCcnt = 4

  for (int i = 0; i < nk; ++i) {
    const int cur = i & 1;
    if (i + 1 < nk) {
      fill(1 - cur, i + 1);             // next tile in flight during compute
      wait_asynccnt<4>();               // current tile's 4 loads retired (in-order)
    } else {
      wait_asynccnt<0>();
    }
    __syncthreads();                    // all waves' current tile visible in LDS

    const float* xcur = Xf + cur * (32 * 132);
    // B fragment: column = wave*16 + lm, elem j -> kk = j + hf*16
    FragBF bf;
    {
      const float* xb = xcur + (size_t)(hf * 16) * 132 + wave * 16 + lm;
#pragma unroll
      for (int g = 0; g < 8; ++g) {
        float lo = xb[(2 * g) * 132];
        float hi = xb[(2 * g + 1) * 132];
        bf.e[2 * g]     = (__bf16)lo;
        bf.e[2 * g + 1] = (__bf16)hi;
      }
    }
    const int ks = i * 32;
#pragma unroll
    for (int m = 0; m < 4; ++m) {
      // A fragment: row = m*16 + lm, elem j -> kk = j + hf*16
      FragBF af;
      const __bf16* wrow = &Wl[(m * 16 + lm) * 196];
#pragma unroll
      for (int g = 0; g < 8; ++g) {
        int kk = ks + g * 2 + hf * 16;
        af.u[g] = *(const unsigned*)&wrow[kk];
      }
      acc[m].v = __builtin_amdgcn_wmma_f32_16x16x32_bf16(
          false, af.v, false, bf.v, (short)0, acc[m].v, false, false);
    }
    __syncthreads();                    // reads of 'cur' done before it is refilled
  }

  // D layout: lane col = lm, row = r + hf*8
#pragma unroll
  for (int m = 0; m < 4; ++m)
#pragma unroll
    for (int r = 0; r < 8; ++r) {
      int row = mtile * 64 + m * 16 + r + hf * 8;
      Yb[(size_t)row * N + ntile * 128 + wave * 16 + lm] = acc[m].e[r];
    }
}

// ---------------------------------------------------------------------------
// Depthwise 3x3 SAME conv (cross-correlation), f32. One thread per pixel.
// ---------------------------------------------------------------------------
__global__ __launch_bounds__(256)
void dwconv3x3(const float* __restrict__ X, const float* __restrict__ Wd,
               float* __restrict__ Y, int C) {
  size_t idx = (size_t)blockIdx.x * 256 + threadIdx.x;
  int x = (int)(idx & 127);
  int y = (int)((idx >> 7) & 127);
  size_t plane = idx >> 14;            // b*C + c
  int c = (int)(plane % C);
  const float* Xp = X + (plane << 14);
  const float* w  = Wd + c * 9;
  float s = 0.f;
#pragma unroll
  for (int ky = 0; ky < 3; ++ky) {
    int yy = y + ky - 1;
    if (yy < 0 || yy > 127) continue;
#pragma unroll
    for (int kx = 0; kx < 3; ++kx) {
      int xx = x + kx - 1;
      if (xx < 0 || xx > 127) continue;
      s += w[ky * 3 + kx] * Xp[yy * 128 + xx];
    }
  }
  Y[idx] = s;
}

// ---------------------------------------------------------------------------
// Per-channel inverse L2 norm over N (matches F.normalize eps semantics).
// ---------------------------------------------------------------------------
__global__ __launch_bounds__(256)
void rownorm(const float* __restrict__ X, float* __restrict__ inv,
             int chanPerB, int chanStrideB) {
  int row = blockIdx.x;
  int b = row / chanPerB, c = row - b * chanPerB;
  const float* p = X + ((size_t)b * chanStrideB + c) * NSP;
  float s = 0.f;
  for (int i = threadIdx.x; i < NSP; i += 256) {
    float v = p[i];
    s += v * v;
  }
  __shared__ float red[256];
  red[threadIdx.x] = s;
  __syncthreads();
  for (int off = 128; off > 0; off >>= 1) {
    if (threadIdx.x < off) red[threadIdx.x] += red[threadIdx.x + off];
    __syncthreads();
  }
  if (threadIdx.x == 0) inv[row] = 1.f / fmaxf(sqrtf(red[0]), 1e-12f);
}

// ---------------------------------------------------------------------------
// S_partial = Q * K^T over a 256-wide K-chunk per wave. 48x48 per (b,h) as
// 3x3 WMMA tiles. Partials written non-atomically -> deterministic reduce.
// grid: (16 k-splits, 32 bh), block 128 (4 waves).
// ---------------------------------------------------------------------------
__global__ __launch_bounds__(128)
void attn_qk(const float* __restrict__ Qd, const float* __restrict__ KVd,
             float* __restrict__ Spart) {
  const int bh = blockIdx.y;
  const int b = bh >> 2, h = bh & 3;
  const int tid = threadIdx.x;
  const int lane = tid & 31, wave = tid >> 5;
  const int lm = lane & 15, hf = lane >> 4;

  const float* Qb = Qd  + ((size_t)b * CCH + (size_t)h * 48) * NSP;
  const float* Kb = KVd + ((size_t)b * 2 * CCH + (size_t)h * 48) * NSP;  // k = first C chans

  FragF acc[3][3];
#pragma unroll
  for (int i = 0; i < 3; ++i)
#pragma unroll
    for (int j = 0; j < 3; ++j)
#pragma unroll
      for (int r = 0; r < 8; ++r) acc[i][j].e[r] = 0.f;

  const int n0 = blockIdx.x * 1024 + wave * 256;
  for (int ks = 0; ks < 256; ks += 32) {
    const int nb = n0 + ks;
    FragBF aq[3], bk[3];
#pragma unroll
    for (int t = 0; t < 3; ++t) {  // A = Q rows (d), K dim = spatial n
      const float* qr = Qb + (size_t)(t * 16 + lm) * NSP + nb + hf * 16;
#pragma unroll
      for (int j = 0; j < 8; ++j) {
        aq[t].e[j]     = (__bf16)qr[j];
        aq[t].e[j + 8] = (__bf16)qr[8 + j];
      }
    }
#pragma unroll
    for (int t = 0; t < 3; ++t) {  // B = K^T: col e = lm, kk = j + hf*16
      const float* kr = Kb + (size_t)(t * 16 + lm) * NSP + nb + hf * 16;
#pragma unroll
      for (int j = 0; j < 16; ++j) bk[t].e[j] = (__bf16)kr[j];
    }
#pragma unroll
    for (int i = 0; i < 3; ++i)
#pragma unroll
      for (int j = 0; j < 3; ++j)
        acc[i][j].v = __builtin_amdgcn_wmma_f32_16x16x32_bf16(
            false, aq[i].v, false, bk[j].v, (short)0, acc[i][j].v, false, false);
  }

  float* Sp = Spart + (size_t)(blockIdx.x * 4 + wave) * (32 * 2304) + (size_t)bh * 2304;
#pragma unroll
  for (int i = 0; i < 3; ++i)
#pragma unroll
    for (int j = 0; j < 3; ++j)
#pragma unroll
      for (int r = 0; r < 8; ++r)
        Sp[(i * 16 + r + hf * 8) * 48 + j * 16 + lm] = acc[i][j].e[r];
}

__global__ __launch_bounds__(256)
void reduce_S(const float* __restrict__ Sp, float* __restrict__ S) {
  int i = blockIdx.x * 256 + threadIdx.x;
  if (i >= 32 * 2304) return;
  float s = 0.f;
  for (int p = 0; p < 64; ++p) s += Sp[(size_t)p * (32 * 2304) + i];
  S[i] = s;
}

// ---------------------------------------------------------------------------
// Scale by invq*invk*temperature, softmax over e (48), write zero-padded
// attn [bh][48][64] (pad so the AV GEMM has K multiple of 32).
// ---------------------------------------------------------------------------
__global__ __launch_bounds__(256)
void softmax_attn(const float* __restrict__ S, const float* __restrict__ invq,
                  const float* __restrict__ invk, const float* __restrict__ temp,
                  float* __restrict__ Ap) {
  int idx = blockIdx.x * 256 + threadIdx.x;
  if (idx >= 32 * 48) return;
  int bh = idx / 48, d = idx - bh * 48;
  int b = bh >> 2, h = bh & 3;
  const float* Sr  = S + (size_t)bh * 2304 + d * 48;
  const float* ikr = invk + (b * CCH + h * 48);
  float iqt = invq[b * CCH + h * 48 + d] * temp[h];
  float mx = -3.4e38f;
  for (int e = 0; e < 48; ++e) mx = fmaxf(mx, Sr[e] * iqt * ikr[e]);
  float sum = 0.f;
  for (int e = 0; e < 48; ++e) sum += __expf(Sr[e] * iqt * ikr[e] - mx);
  float inv = 1.f / sum;
  float* Ar = Ap + (size_t)bh * 48 * 64 + d * 64;
  for (int e = 0; e < 48; ++e) Ar[e] = __expf(Sr[e] * iqt * ikr[e] - mx) * inv;
  for (int e = 48; e < 64; ++e) Ar[e] = 0.f;
}

// ---------------------------------------------------------------------------
// out[b,h,d,n] = sum_e attn[d,e] * V[e,n].  A (attn, 48x64 padded) in regs,
// B = V columns, 3 M-tiles x 2 K-steps of WMMA per 16-col block per wave.
// grid: (16384/128, 32), block 256.
// ---------------------------------------------------------------------------
__global__ __launch_bounds__(256)
void attn_av(const float* __restrict__ Ap, const float* __restrict__ KVd,
             float* __restrict__ O) {
  const int bh = blockIdx.y;
  const int b = bh >> 2, h = bh & 3;
  const int tid = threadIdx.x;
  const int lane = tid & 31, wave = tid >> 5;
  const int lm = lane & 15, hf = lane >> 4;
  const int n0 = blockIdx.x * 128 + wave * 16;

  const float* Ab = Ap + (size_t)bh * 48 * 64;
  const float* Vb = KVd + ((size_t)b * 2 * CCH + CCH + (size_t)h * 48) * NSP;  // v = second C chans

  FragBF af[3][2];
#pragma unroll
  for (int mt = 0; mt < 3; ++mt)
#pragma unroll
    for (int ks = 0; ks < 2; ++ks) {
      const float* ar = Ab + (size_t)(mt * 16 + lm) * 64 + ks * 32;
#pragma unroll
      for (int j = 0; j < 8; ++j) {
        af[mt][ks].e[j]     = (__bf16)ar[j + hf * 8];
        af[mt][ks].e[j + 8] = (__bf16)ar[16 + j + hf * 8];
      }
    }

  FragBF bv[2];
#pragma unroll
  for (int ks = 0; ks < 2; ++ks)
#pragma unroll
    for (int j = 0; j < 16; ++j) {
      int e = ks * 32 + (j & 7) + (j >> 3) * 16 + hf * 8;
      bv[ks].e[j] = (e < 48) ? (__bf16)Vb[(size_t)e * NSP + n0 + lm] : (__bf16)0.f;
    }

#pragma unroll
  for (int mt = 0; mt < 3; ++mt) {
    FragF acc;
#pragma unroll
    for (int r = 0; r < 8; ++r) acc.e[r] = 0.f;
    acc.v = __builtin_amdgcn_wmma_f32_16x16x32_bf16(
        false, af[mt][0].v, false, bv[0].v, (short)0, acc.v, false, false);
    acc.v = __builtin_amdgcn_wmma_f32_16x16x32_bf16(
        false, af[mt][1].v, false, bv[1].v, (short)0, acc.v, false, false);
#pragma unroll
    for (int r = 0; r < 8; ++r) {
      int row = h * 48 + mt * 16 + r + hf * 8;
      O[((size_t)b * CCH + row) * NSP + n0 + lm] = acc.e[r];
    }
  }
}

// ---------------------------------------------------------------------------
extern "C" void kernel_launch(void* const* d_in, const int* in_sizes, int n_in,
                              void* d_out, int out_size, void* d_ws, size_t ws_size,
                              hipStream_t stream) {
  const float* f_opt  = (const float*)d_in[0];
  const float* f_sar  = (const float*)d_in[1];
  const float* w_q    = (const float*)d_in[2];
  const float* w_qdw  = (const float*)d_in[3];
  const float* w_kv   = (const float*)d_in[4];
  const float* w_kvdw = (const float*)d_in[5];
  const float* w_out  = (const float*)d_in[6];
  const float* temp   = (const float*)d_in[7];
  float* out = (float*)d_out;

  float* ws = (float*)d_ws;
  const size_t QN  = (size_t)NB * CCH * NSP;       // 25,165,824
  const size_t KVN = (size_t)NB * 2 * CCH * NSP;   // 50,331,648
  float* q1    = ws;                 // q after 1x1; later reused as attn output
  float* qd    = q1 + QN;            // q after depthwise
  float* kv1   = qd + QN;            // kv after 1x1
  float* kvd   = kv1 + KVN;          // kv after depthwise (k | v)
  float* invq  = kvd + KVN;          // 1536
  float* invk  = invq + 1536;        // 1536
  float* S     = invk + 1536;        // 32*48*48
  float* Apad  = S + 32 * 2304;      // 32*48*64
  float* Spart = Apad + 32 * 48 * 64;// 64 * 32*2304 partials

  dim3 b256(256);

  // q = dwconv(conv1x1(f_opt))
  pw_conv_bf16<<<dim3(128, 3, 8), b256, 0, stream>>>(f_opt, w_q, q1, CCH, NSP);
  dwconv3x3<<<dim3(98304), b256, 0, stream>>>(q1, w_qdw, qd, CCH);

  // kv = dwconv(conv1x1(f_sar))
  pw_conv_bf16<<<dim3(128, 6, 8), b256, 0, stream>>>(f_sar, w_kv, kv1, CCH, NSP);
  dwconv3x3<<<dim3(196608), b256, 0, stream>>>(kv1, w_kvdw, kvd, 2 * CCH);

  // inverse L2 norms (folded into softmax scaling)
  rownorm<<<dim3(1536), b256, 0, stream>>>(qd, invq, CCH, CCH);
  rownorm<<<dim3(1536), b256, 0, stream>>>(kvd, invk, CCH, 2 * CCH);

  // S = Q K^T (split-K partials, deterministic reduce)
  attn_qk<<<dim3(16, 32), dim3(128), 0, stream>>>(qd, kvd, Spart);
  reduce_S<<<dim3((32 * 2304 + 255) / 256), b256, 0, stream>>>(Spart, S);

  // softmax(scale * S) -> padded attn
  softmax_attn<<<dim3(6), b256, 0, stream>>>(S, invq, invk, temp, Apad);

  // out_attn = attn @ V  (reuse q1 buffer)
  attn_av<<<dim3(128, 32), b256, 0, stream>>>(Apad, kvd, q1);

  // final 1x1 projection -> d_out
  pw_conv_bf16<<<dim3(128, 3, 8), b256, 0, stream>>>(q1, w_out, out, CCH, NSP);
}
